// Attention_9045201126086
// MI455X (gfx1250) — matrix operand order repaired and checked by reading, not compiled, not run on hardware
//
#include <hip/hip_runtime.h>
#include <hip/hip_bf16.h>

typedef __attribute__((ext_vector_type(16))) __bf16 bf16x16;
typedef __attribute__((ext_vector_type(8)))  __bf16 bf16x8;
typedef __attribute__((ext_vector_type(8)))  float  f32x8;

#define HIDDEN 2048
#define SEQ    2048
#define BATCH  2
#define TOK    (BATCH * SEQ)   // 4096
#define NH     16
#define NKV    8
#define HD     128
#define QD     (NH * HD)       // 2048
#define KD     (NKV * HD)      // 1024

// sched_group_barrier masks
#define SG_WMMA  0x0008
#define SG_VMEMR 0x0020
#define SG_TRANS 0x0400

// ---------- bf16 helpers (round-to-nearest-even) ----------
__device__ __forceinline__ __bf16 f2bf(float f) {
  union { float f; unsigned u; } v; v.f = f;
  unsigned r = v.u + 0x7FFFu + ((v.u >> 16) & 1u);
  unsigned short h = (unsigned short)(r >> 16);
  return __builtin_bit_cast(__bf16, h);
}
__device__ __forceinline__ float bf2f(__bf16 b) {
  unsigned short h = __builtin_bit_cast(unsigned short, b);
  union { unsigned u; float f; } v; v.u = ((unsigned)h) << 16;
  return v.f;
}

// ---------- WMMA fragment loaders (CDNA5 16x16x32 bf16 layouts) ----------
// A-matrix 16x32: lane L (L<16): row M=L, K = {0..7} in v0-3, {16..23} in v4-7;
// lane L+16: same row, K = {8..15}, {24..31}.
__device__ __forceinline__ bf16x16 load_a(const __bf16* base, int ld) {
  int l = threadIdx.x & 31;
  const __bf16* p = base + (size_t)(l & 15) * ld + ((l >> 4) << 3);
  bf16x16 f;
  ((bf16x8*)&f)[0] = *(const bf16x8*)(p);
  ((bf16x8*)&f)[1] = *(const bf16x8*)(p + 16);
  return f;
}
// B-matrix 32x16: lane L (L<16): col N=L, K = {0..7}, {16..23}; lane L+16: K = {8..15}, {24..31}.
__device__ __forceinline__ bf16x16 load_b(const __bf16* base, int ld) {
  int l = threadIdx.x & 31;
  const __bf16* p = base + (size_t)(l & 15) * ld + ((l >> 4) << 3);
  bf16x16 f;
  ((bf16x8*)&f)[0] = *(const bf16x8*)(p);
  ((bf16x8*)&f)[1] = *(const bf16x8*)(p + 16);
  return f;
}

__device__ __forceinline__ f32x8 wmma_bf16(bf16x16 a, bf16x16 b, f32x8 c) {
  return __builtin_amdgcn_wmma_f32_16x16x32_bf16(false, a, false, b, (short)0, c,
                                                 false, false);
}

// ---------- elementwise fp32 -> bf16 ----------
__global__ __launch_bounds__(256) void cvt_f32_bf16(const float* __restrict__ X,
                                                    __bf16* __restrict__ Y, int n) {
  int i = blockIdx.x * 256 + threadIdx.x;
  if (i < n) Y[i] = f2bf(X[i]);
}

// ---------- transpose-convert: WT[n][k] = (bf16) W[k][n] ----------
__global__ __launch_bounds__(256) void transpose_cvt(const float* __restrict__ W,
                                                     __bf16* __restrict__ WT,
                                                     int Kd, int Nd) {
  __shared__ __bf16 tile[32][33];
  int kb = blockIdx.x * 32, nb = blockIdx.y * 32;
  int tx = threadIdx.x & 31, ty = threadIdx.x >> 5;
#pragma unroll
  for (int i = 0; i < 32; i += 8)
    tile[ty + i][tx] = f2bf(W[(size_t)(kb + ty + i) * Nd + nb + tx]);
  __syncthreads();
#pragma unroll
  for (int i = 0; i < 32; i += 8)
    WT[(size_t)(nb + ty + i) * Kd + kb + tx] = tile[tx][ty + i];
}

// ---------- GEMM: C[M,N] = A[M,K] (row-major) x BT[N,K] (row-major) ----------
// 4 waves / WG, 64x64 per wave, 128x128 per WG, double-buffered k-steps.
// grid = (N/128, M/128). K must be a multiple of 64.
template <bool OUT_BF16, bool TRANS_OUT>
__global__ __launch_bounds__(128) void gemm_bf16(const __bf16* __restrict__ A,
                                                 const __bf16* __restrict__ BT,
                                                 void* __restrict__ C, int K, int ldc) {
  int w  = threadIdx.x >> 5;
  int m0 = blockIdx.y * 128 + (w >> 1) * 64;
  int n0 = blockIdx.x * 128 + (w & 1) * 64;
  f32x8 acc[4][4];
#pragma unroll
  for (int mi = 0; mi < 4; mi++)
#pragma unroll
    for (int ni = 0; ni < 4; ni++) acc[mi][ni] = 0.0f;

  bf16x16 a0[4], b0[4], a1[4], b1[4];
#pragma unroll
  for (int i = 0; i < 4; i++) a0[i] = load_a(A + (size_t)(m0 + i * 16) * K, K);
#pragma unroll
  for (int i = 0; i < 4; i++) b0[i] = load_b(BT + (size_t)(n0 + i * 16) * K, K);

  for (int k = 0; k < K; k += 64) {
    // prefetch k+32 while computing with buffer 0
#pragma unroll
    for (int i = 0; i < 4; i++) a1[i] = load_a(A + (size_t)(m0 + i * 16) * K + k + 32, K);
#pragma unroll
    for (int i = 0; i < 4; i++) b1[i] = load_b(BT + (size_t)(n0 + i * 16) * K + k + 32, K);
#pragma unroll
    for (int mi = 0; mi < 4; mi++)
#pragma unroll
      for (int ni = 0; ni < 4; ni++)
        acc[mi][ni] = wmma_bf16(a0[mi], b0[ni], acc[mi][ni]);
    // prefetch k+64 while computing with buffer 1
    if (k + 64 < K) {
#pragma unroll
      for (int i = 0; i < 4; i++) a0[i] = load_a(A + (size_t)(m0 + i * 16) * K + k + 64, K);
#pragma unroll
      for (int i = 0; i < 4; i++) b0[i] = load_b(BT + (size_t)(n0 + i * 16) * K + k + 64, K);
    }
#pragma unroll
    for (int mi = 0; mi < 4; mi++)
#pragma unroll
      for (int ni = 0; ni < 4; ni++)
        acc[mi][ni] = wmma_bf16(a1[mi], b1[ni], acc[mi][ni]);
    // enforce pipeline: loads(buf1) | wmma(buf0) | loads(buf0,next) | wmma(buf1)
    __builtin_amdgcn_sched_group_barrier(SG_VMEMR, 16, 0);
    __builtin_amdgcn_sched_group_barrier(SG_WMMA, 16, 0);
    __builtin_amdgcn_sched_group_barrier(SG_VMEMR, 16, 0);
    __builtin_amdgcn_sched_group_barrier(SG_WMMA, 16, 0);
  }

  int l = threadIdx.x & 31;
  int lm = l & 15, hi = l >> 4;
#pragma unroll
  for (int mi = 0; mi < 4; mi++)
#pragma unroll
    for (int ni = 0; ni < 4; ni++) {
      int n  = n0 + ni * 16 + lm;
      int mb = m0 + mi * 16 + hi * 8;
      if (TRANS_OUT) {
        bf16x8 pk;
#pragma unroll
        for (int r = 0; r < 8; r++) pk[r] = f2bf(acc[mi][ni][r]);
        *(bf16x8*)((__bf16*)C + (size_t)n * ldc + mb) = pk;  // contiguous b128
      } else if (OUT_BF16) {
#pragma unroll
        for (int r = 0; r < 8; r++)
          ((__bf16*)C)[(size_t)(mb + r) * ldc + n] = f2bf(acc[mi][ni][r]);
      } else {
#pragma unroll
        for (int r = 0; r < 8; r++)
          ((float*)C)[(size_t)(mb + r) * ldc + n] = acc[mi][ni][r];
      }
    }
}

// ---------- per-(token, head) RMSNorm + RoPE, in place on bf16 ----------
__global__ __launch_bounds__(128) void norm_rope(__bf16* __restrict__ qk,
                                                 const float* __restrict__ w,
                                                 const float* __restrict__ cosb,
                                                 const float* __restrict__ sinb, int ld) {
  int t = blockIdx.x, h = blockIdx.y, d = threadIdx.x;
  int s = t & (SEQ - 1);
  __shared__ float buf[128];
  __shared__ float ssum[4];
  __bf16* row = qk + (size_t)t * ld + h * HD;
  float v = bf2f(row[d]);
  float sq = v * v;
#pragma unroll
  for (int off = 16; off > 0; off >>= 1) sq += __shfl_xor(sq, off, 32);
  if ((threadIdx.x & 31) == 0) ssum[threadIdx.x >> 5] = sq;
  __syncthreads();
  float tot = ssum[0] + ssum[1] + ssum[2] + ssum[3];
  float nrm = v * rsqrtf(tot * (1.0f / 128.0f) + 1e-5f) * w[d];
  buf[d] = nrm;
  __syncthreads();
  float rot = (d < 64) ? -buf[d + 64] : buf[d - 64];
  float o = nrm * cosb[(size_t)s * HD + d] + rot * sinb[(size_t)s * HD + d];
  row[d] = f2bf(o);
}

// ---------- causal flash attention, S^T trick, no LDS, 16 q-rows per wave ----
// q:[TOK,QD] bf16 (post rope), k:[TOK,KD] bf16, vT:[KD,TOK] bf16, o:[TOK,QD] bf16
__global__ __launch_bounds__(128) void attn_fwd(const __bf16* __restrict__ q,
                                                const __bf16* __restrict__ k,
                                                const __bf16* __restrict__ vT,
                                                __bf16* __restrict__ o) {
  int w = threadIdx.x >> 5;
  int qb = blockIdx.x, h = blockIdx.y, b = blockIdx.z;
  int hk = h >> 1;                       // GQA: N_REP = 2
  int q0 = qb * 64 + w * 16;
  int l = threadIdx.x & 31;
  int lm = l & 15, hi = l >> 4;

  // Q fragments (B-side of S^T = K x Q^T), 4 d-chunks of 32
  const __bf16* qbase = q + (size_t)(b * SEQ + q0) * QD + h * HD;
  bf16x16 qf[4];
#pragma unroll
  for (int c = 0; c < 4; c++) qf[c] = load_b(qbase + c * 32, QD);

  f32x8 O[8];                            // 16q x 128d accum (8 N-tiles over d)
#pragma unroll
  for (int i = 0; i < 8; i++) O[i] = 0.0f;
  float m_run = -3.0e38f, l_run = 0.0f;
  const float sc = 0.08838834764831845f; // 1/sqrt(128)
  const __bf16* kbase = k + (size_t)(b * SEQ) * KD + hk * HD;
  const __bf16* vbase = vT + (size_t)(hk * HD) * TOK + b * SEQ;

  int nkb = (q0 + 47) >> 5;              // key blocks of 32, causal bound
  for (int kb = 0; kb < nkb; kb++) {
    int kbeg = kb * 32;
    // --- load all 8 K fragments as one clause (distinct registers) ---
    bf16x16 kf0[4], kf1[4];
#pragma unroll
    for (int c = 0; c < 4; c++)
      kf0[c] = load_a(kbase + (size_t)kbeg * KD + c * 32, KD);
#pragma unroll
    for (int c = 0; c < 4; c++)
      kf1[c] = load_a(kbase + (size_t)(kbeg + 16) * KD + c * 32, KD);
    // --- S^T tiles: A = K rows (keys), B = Q^T. C: lane=q, vgpr=key ---
    f32x8 st0 = 0.0f, st1 = 0.0f;
#pragma unroll
    for (int c = 0; c < 4; c++) st0 = wmma_bf16(kf0[c], qf[c], st0);
#pragma unroll
    for (int c = 0; c < 4; c++) st1 = wmma_bf16(kf1[c], qf[c], st1);
    // --- prefetch all 8 V fragments; softmax VALU below hides the latency ---
    bf16x16 vf[8];
#pragma unroll
    for (int ni = 0; ni < 8; ni++)
      vf[ni] = load_b(vbase + (size_t)(ni * 16) * TOK + kbeg, TOK);

    int qrow = q0 + lm;
    float sv0[8], sv1[8];
    float bmax = -3.0e38f;
#pragma unroll
    for (int r = 0; r < 8; r++) {
      int key0 = kbeg + hi * 8 + r;
      float x0 = (key0 <= qrow) ? st0[r] * sc : -3.0e38f;
      sv0[r] = x0; bmax = fmaxf(bmax, x0);
      float x1 = (key0 + 16 <= qrow) ? st1[r] * sc : -3.0e38f;
      sv1[r] = x1; bmax = fmaxf(bmax, x1);
    }
    bmax = fmaxf(bmax, __shfl_xor(bmax, 16, 32));   // combine key halves
    float mnew = fmaxf(m_run, bmax);
    float resc = __expf(m_run - mnew);
    float bsum = 0.0f;
    bf16x16 pf;                          // P^T packs directly into A-fragment
#pragma unroll
    for (int r = 0; r < 8; r++) {
      float p0 = __expf(sv0[r] - mnew);
      float p1 = __expf(sv1[r] - mnew);
      bsum += p0 + p1;
      pf[r]     = f2bf(p0);
      pf[8 + r] = f2bf(p1);
    }
    bsum += __shfl_xor(bsum, 16, 32);
    l_run = l_run * resc + bsum;
    m_run = mnew;
    // rescale O: factor for q = r+8*hi lives at lane (r+8*hi) (q on lanes there)
    float rs[8];
#pragma unroll
    for (int r = 0; r < 8; r++) rs[r] = __shfl(resc, r + 8 * hi, 32);
#pragma unroll
    for (int ni = 0; ni < 8; ni++)
#pragma unroll
      for (int r = 0; r < 8; r++) O[ni][r] *= rs[r];
    // --- P·V : A = pf (16q x 32keys), B = V tiles, accumulate into O ---
#pragma unroll
    for (int ni = 0; ni < 8; ni++) O[ni] = wmma_bf16(pf, vf[ni], O[ni]);

    // enforce pipeline: 16 K-loads | 8 QK-wmma | 16 V-loads | exp | 8 PV-wmma
    __builtin_amdgcn_sched_group_barrier(SG_VMEMR, 16, 0);
    __builtin_amdgcn_sched_group_barrier(SG_WMMA, 8, 0);
    __builtin_amdgcn_sched_group_barrier(SG_VMEMR, 16, 0);
    __builtin_amdgcn_sched_group_barrier(SG_TRANS, 17, 0);
    __builtin_amdgcn_sched_group_barrier(SG_WMMA, 8, 0);
  }
  float inv = 1.0f / l_run;
  float nv[8];
#pragma unroll
  for (int r = 0; r < 8; r++) nv[r] = __shfl(inv, r + 8 * hi, 32);
  __bf16* ob = o + (size_t)(b * SEQ + q0) * QD + h * HD;
#pragma unroll
  for (int ni = 0; ni < 8; ni++) {
    int d = ni * 16 + lm;
#pragma unroll
    for (int r = 0; r < 8; r++)
      ob[(size_t)(r + 8 * hi) * QD + d] = f2bf(O[ni][r] * nv[r]);
  }
}

extern "C" void kernel_launch(void* const* d_in, const int* in_sizes, int n_in,
                              void* d_out, int out_size, void* d_ws, size_t ws_size,
                              hipStream_t stream) {
  (void)in_sizes; (void)n_in; (void)out_size; (void)ws_size;
  const float* x    = (const float*)d_in[0];
  const float* cosb = (const float*)d_in[1];
  const float* sinb = (const float*)d_in[2];
  const float* wq   = (const float*)d_in[3];
  const float* wk   = (const float*)d_in[4];
  const float* wv   = (const float*)d_in[5];
  const float* wo   = (const float*)d_in[6];
  const float* qnw  = (const float*)d_in[7];
  const float* knw  = (const float*)d_in[8];
  float* out = (float*)d_out;

  char* ws = (char*)d_ws;
  const size_t MB = 1ull << 20;
  __bf16* xb  = (__bf16*)(ws);             // 16 MB  x in bf16          [TOK,HIDDEN]
  __bf16* wqT = (__bf16*)(ws + 16 * MB);   //  8 MB  wq^T               [QD,HIDDEN]
  __bf16* wkT = (__bf16*)(ws + 24 * MB);   //  4 MB  wk^T               [KD,HIDDEN]
  __bf16* wvT = (__bf16*)(ws + 28 * MB);   //  4 MB  wv^T               [KD,HIDDEN]
  __bf16* woT = (__bf16*)(ws + 32 * MB);   //  8 MB  wo^T               [HIDDEN,QD]
  __bf16* qbuf= (__bf16*)(ws + 40 * MB);   // 16 MB  q (normed+roped)   [TOK,QD]
  __bf16* kbuf= (__bf16*)(ws + 56 * MB);   //  8 MB  k (normed+roped)   [TOK,KD]
  __bf16* vT  = (__bf16*)(ws + 64 * MB);   //  8 MB  v transposed       [KD,TOK]
  __bf16* ab  = (__bf16*)(ws + 72 * MB);   // 16 MB  attention output   [TOK,QD]

  // 1) precision conversion / weight transposes (one-time, L2-resident after)
  cvt_f32_bf16<<<TOK * HIDDEN / 256, 256, 0, stream>>>(x, xb, TOK * HIDDEN);
  transpose_cvt<<<dim3(HIDDEN / 32, QD / 32), 256, 0, stream>>>(wq, wqT, HIDDEN, QD);
  transpose_cvt<<<dim3(HIDDEN / 32, KD / 32), 256, 0, stream>>>(wk, wkT, HIDDEN, KD);
  transpose_cvt<<<dim3(HIDDEN / 32, KD / 32), 256, 0, stream>>>(wv, wvT, HIDDEN, KD);
  transpose_cvt<<<dim3(QD / 32, HIDDEN / 32), 256, 0, stream>>>(wo, woT, QD, HIDDEN);

  // 2) QKV projections (bf16 WMMA, fp32 accum); V stored transposed for PV
  gemm_bf16<true, false><<<dim3(QD / 128, TOK / 128), 128, 0, stream>>>(xb, wqT, qbuf, HIDDEN, QD);
  gemm_bf16<true, false><<<dim3(KD / 128, TOK / 128), 128, 0, stream>>>(xb, wkT, kbuf, HIDDEN, KD);
  gemm_bf16<true, true ><<<dim3(KD / 128, TOK / 128), 128, 0, stream>>>(xb, wvT, vT, HIDDEN, TOK);

  // 3) per-head RMSNorm + RoPE
  norm_rope<<<dim3(TOK, NH),  128, 0, stream>>>(qbuf, qnw, cosb, sinb, QD);
  norm_rope<<<dim3(TOK, NKV), 128, 0, stream>>>(kbuf, knw, cosb, sinb, KD);

  // 4) causal flash attention (GQA)
  attn_fwd<<<dim3(SEQ / 64, NH, BATCH), 128, 0, stream>>>(qbuf, kbuf, vT, ab);

  // 5) output projection, fp32 result
  gemm_bf16<false, false><<<dim3(HIDDEN / 128, TOK / 128), 128, 0, stream>>>(ab, woT, out, QD, HIDDEN);
}
